// AttendCRNN_67817533604106
// MI455X (gfx1250) — matrix operand order repaired and checked by reading, not compiled, not run on hardware
//
#include <hip/hip_runtime.h>
#include <hip/hip_bf16.h>
#include <stdint.h>

// ---------------------------------------------------------------------------
// Types / helpers
// ---------------------------------------------------------------------------
typedef __bf16 bf16;
typedef __attribute__((ext_vector_type(16))) __bf16 v16bf;
typedef __attribute__((ext_vector_type(8)))  __bf16 v8bf;
typedef __attribute__((ext_vector_type(8)))  float  v8f;

static __device__ __forceinline__ bf16 us2bf(unsigned short u) {
    return __builtin_bit_cast(bf16, u);
}
static __device__ __forceinline__ float bf2f(unsigned short u) {
    unsigned x = ((unsigned)u) << 16;
    return __builtin_bit_cast(float, x);
}
static __device__ __forceinline__ unsigned short f2bf(float f) {
    unsigned u = __builtin_bit_cast(unsigned, f);
    unsigned r = u + 0x7FFFu + ((u >> 16) & 1u);   // round-to-nearest-even
    return (unsigned short)(r >> 16);
}
static __device__ __forceinline__ float lrelu(float v) { return v >= 0.f ? v : 0.2f * v; }
static __device__ __forceinline__ float sigm(float x) { return 1.0f / (1.0f + expf(-x)); }

static __device__ __forceinline__ v16bf zero16bf() {
    v16bf z;
    #pragma unroll
    for (int i = 0; i < 16; ++i) z[i] = us2bf(0);
    return z;
}

// A-tile loader: 16x32 bf16 tile, row fixed per lane, lda in elements.
// lane half lh: K = lh*16 + {0..15}, contiguous per lane.
static __device__ __forceinline__ v16bf loadA(const unsigned short* __restrict__ base,
                                              long row, int lda, int k0, int lh) {
    const unsigned short* pa = base + row * lda + k0 + (lh << 4);
    v8bf lo = *(const v8bf*)pa;
    v8bf hi = *(const v8bf*)(pa + 8);
    v16bf a;
    #pragma unroll
    for (int i = 0; i < 8; ++i) { a[i] = lo[i]; a[i + 8] = hi[i]; }
    return a;
}
// B-tile loader, layout0 (B(k,n) = base[n*ldb + k]): lanes 0-15 K=k0..k0+15,
// lanes 16-31 K=k0+16..k0+31, contiguous per lane.
static __device__ __forceinline__ v16bf loadB0(const unsigned short* __restrict__ base,
                                               long col, int ldb, int k0, int lh) {
    const unsigned short* pb = base + col * ldb + k0 + (lh << 4);
    v8bf lo = *(const v8bf*)pb;
    v8bf hi = *(const v8bf*)(pb + 8);
    v16bf b;
    #pragma unroll
    for (int i = 0; i < 8; ++i) { b[i] = lo[i]; b[i + 8] = hi[i]; }
    return b;
}

// ---------------------------------------------------------------------------
// Weight repack / convert kernels
// ---------------------------------------------------------------------------
__global__ void k_conv_repack(const float* __restrict__ w, unsigned short* __restrict__ out,
                              int Cout, int Cin, int KH, int KW) {
    long total = (long)Cout * Cin * KH * KW;
    long i = (long)blockIdx.x * blockDim.x + threadIdx.x;
    if (i >= total) return;
    int ci = (int)(i % Cin);
    long t = i / Cin;
    int co = (int)(t % Cout);
    int kk = (int)(t / Cout);
    int kh = kk / KW, kw = kk % KW;
    out[i] = f2bf(w[(((long)co * Cin + ci) * KH + kh) * KW + kw]);
}

__global__ void k_f32_to_bf16(const float* __restrict__ in, unsigned short* __restrict__ out, long n) {
    long i = (long)blockIdx.x * blockDim.x + threadIdx.x;
    if (i < n) out[i] = f2bf(in[i]);
}

__global__ void k_add_bias(const float* __restrict__ a, const float* __restrict__ b,
                           float* __restrict__ out, int n) {
    int i = blockIdx.x * blockDim.x + threadIdx.x;
    if (i < n) out[i] = a[i] + b[i];
}

// ---------------------------------------------------------------------------
// conv1: Cin=1 direct (N=32, 32x512 -> 64ch, 3x3 pad1, lrelu), NHWC bf16 out
// ---------------------------------------------------------------------------
__global__ void k_conv1(const float* __restrict__ x, const float* __restrict__ w,
                        const float* __restrict__ bias, unsigned short* __restrict__ out) {
    long i = (long)blockIdx.x * blockDim.x + threadIdx.x;   // total 32*32*512*64
    if (i >= (long)32 * 32 * 512 * 64) return;
    int co = (int)(i & 63);
    long t = i >> 6;
    int wo = (int)(t & 511); t >>= 9;
    int ho = (int)(t & 31);
    int n  = (int)(t >> 5);
    float acc = bias[co];
    #pragma unroll
    for (int kh = 0; kh < 3; ++kh) {
        int hi = ho + kh - 1;
        if ((unsigned)hi >= 32u) continue;
        #pragma unroll
        for (int kw = 0; kw < 3; ++kw) {
            int wi = wo + kw - 1;
            if ((unsigned)wi >= 512u) continue;
            acc += x[((long)n * 32 + hi) * 512 + wi] * w[co * 9 + kh * 3 + kw];
        }
    }
    out[i] = f2bf(lrelu(acc));
}

// ---------------------------------------------------------------------------
// WMMA implicit-GEMM conv, 2x2 register-blocked (32x32 macro-tile per wave).
// in NHWC bf16, weights (kh,kw,co,ci) bf16, out NHWC bf16.
// M = N*Ho*Wo and Cout are multiples of 32 for every layer here.
// ---------------------------------------------------------------------------
__global__ void k_conv_wmma(const unsigned short* __restrict__ in,
                            const unsigned short* __restrict__ wpk,
                            const float* __restrict__ bias,
                            unsigned short* __restrict__ out,
                            int Nimg, int Hi, int Wi, int Cin,
                            int Ho, int Wo, int Cout,
                            int KH, int KW, int pad, int act) {
    int Nt2 = Cout >> 5;
    int Mt2 = (Nimg * Ho * Wo) >> 5;
    int total = Mt2 * Nt2;
    int wid = blockIdx.x * (blockDim.x >> 5) + (threadIdx.x >> 5);
    if (wid >= total) return;
    int lane = threadIdx.x & 31;
    int lh = lane >> 4;
    int l16 = lane & 15;
    int m0 = (wid / Nt2) << 5;
    int n0 = (wid % Nt2) << 5;

    // two A pixel rows per lane (sub-tiles m0 and m0+16)
    int prow0 = m0 + l16;
    int prow1 = m0 + 16 + l16;
    int wo0 = prow0 % Wo; int t0 = prow0 / Wo; int ho0 = t0 % Ho; int ni0 = t0 / Ho;
    int wo1 = prow1 % Wo; int t1 = prow1 / Wo; int ho1 = t1 % Ho; int ni1 = t1 / Ho;
    int co0 = n0 + l16;
    int co1 = n0 + 16 + l16;

    v8f acc00 = {}, acc01 = {}, acc10 = {}, acc11 = {};
    for (int kh = 0; kh < KH; ++kh) {
        int hi0 = ho0 + kh - pad, hi1 = ho1 + kh - pad;
        for (int kw = 0; kw < KW; ++kw) {
            int wi0 = wo0 + kw - pad, wi1 = wo1 + kw - pad;
            bool ok0 = ((unsigned)hi0 < (unsigned)Hi) && ((unsigned)wi0 < (unsigned)Wi);
            bool ok1 = ((unsigned)hi1 < (unsigned)Hi) && ((unsigned)wi1 < (unsigned)Wi);
            const unsigned short* abase0 = in + ((long)(ni0 * Hi + hi0) * Wi + wi0) * Cin;
            const unsigned short* abase1 = in + ((long)(ni1 * Hi + hi1) * Wi + wi1) * Cin;
            const unsigned short* bbase0 = wpk + ((long)((kh * KW + kw) * Cout + co0)) * Cin;
            const unsigned short* bbase1 = wpk + ((long)((kh * KW + kw) * Cout + co1)) * Cin;
            for (int c0 = 0; c0 < Cin; c0 += 32) {
                v16bf a0, a1;
                if (ok0) { a0 = loadA(abase0, 0, 0, c0, lh); }
                else     { a0 = zero16bf(); }
                if (ok1) { a1 = loadA(abase1, 0, 0, c0, lh); }
                else     { a1 = zero16bf(); }
                v16bf b0 = loadB0(bbase0, 0, 0, c0, lh);
                v16bf b1 = loadB0(bbase1, 0, 0, c0, lh);
                acc00 = __builtin_amdgcn_wmma_f32_16x16x32_bf16(false, a0, false, b0, (short)0, acc00, false, false);
                acc01 = __builtin_amdgcn_wmma_f32_16x16x32_bf16(false, a0, false, b1, (short)0, acc01, false, false);
                acc10 = __builtin_amdgcn_wmma_f32_16x16x32_bf16(false, a1, false, b0, (short)0, acc10, false, false);
                acc11 = __builtin_amdgcn_wmma_f32_16x16x32_bf16(false, a1, false, b1, (short)0, acc11, false, false);
            }
        }
    }
    float bv0 = bias ? bias[co0] : 0.f;
    float bv1 = bias ? bias[co1] : 0.f;
    #pragma unroll
    for (int mi = 0; mi < 2; ++mi) {
        #pragma unroll
        for (int r = 0; r < 8; ++r) {
            int row = m0 + mi * 16 + r + (lh << 3);
            int wo2 = row % Wo; int t2 = row / Wo;
            int ho2 = t2 % Ho;  int n2 = t2 / Ho;
            long obase = ((long)(n2 * Ho + ho2) * Wo + wo2) * Cout;
            float v0 = (mi ? acc10[r] : acc00[r]) + bv0;
            float v1 = (mi ? acc11[r] : acc01[r]) + bv1;
            if (act) { v0 = lrelu(v0); v1 = lrelu(v1); }
            out[obase + co0] = f2bf(v0);
            out[obase + co1] = f2bf(v1);
        }
    }
}

// ---------------------------------------------------------------------------
// MaxPool 2x2, stride (sh,sw), pad (0,pw), NHWC bf16
// ---------------------------------------------------------------------------
__global__ void k_maxpool(const unsigned short* __restrict__ in, unsigned short* __restrict__ out,
                          int N, int Hi, int Wi, int C, int Ho, int Wo,
                          int sh, int sw, int pw) {
    long total = (long)N * Ho * Wo * C;
    long i = (long)blockIdx.x * blockDim.x + threadIdx.x;
    if (i >= total) return;
    int c = (int)(i % C);
    long t = i / C;
    int wo = (int)(t % Wo); t /= Wo;
    int ho = (int)(t % Ho);
    int n  = (int)(t / Ho);
    float m = -3.4e38f;
    #pragma unroll
    for (int kh = 0; kh < 2; ++kh) {
        int hi = ho * sh + kh;
        if (hi >= Hi) continue;
        #pragma unroll
        for (int kw = 0; kw < 2; ++kw) {
            int wi = wo * sw + kw - pw;
            if ((unsigned)wi >= (unsigned)Wi) continue;
            m = fmaxf(m, bf2f(in[((long)(n * Hi + hi) * Wi + wi) * C + c]));
        }
    }
    out[i] = f2bf(m);
}

// ---------------------------------------------------------------------------
// BatchNorm (training-mode batch stats) over NHWC bf16
// ---------------------------------------------------------------------------
__global__ void k_bn_stats(const unsigned short* __restrict__ x, int C, int count,
                           float* __restrict__ stats) {
    int c = blockIdx.x;
    __shared__ float ssum[256], ssq[256];
    float s = 0.f, q = 0.f;
    for (int i = threadIdx.x; i < count; i += 256) {
        float v = bf2f(x[(long)i * C + c]);
        s += v; q += v * v;
    }
    ssum[threadIdx.x] = s; ssq[threadIdx.x] = q;
    __syncthreads();
    for (int off = 128; off > 0; off >>= 1) {
        if (threadIdx.x < off) {
            ssum[threadIdx.x] += ssum[threadIdx.x + off];
            ssq[threadIdx.x]  += ssq[threadIdx.x + off];
        }
        __syncthreads();
    }
    if (threadIdx.x == 0) {
        float mu = ssum[0] / count;
        float var = ssq[0] / count - mu * mu;
        stats[c] = mu;
        stats[C + c] = rsqrtf(var + 1e-5f);
    }
}

__global__ void k_bn_apply(unsigned short* __restrict__ x, const float* __restrict__ stats,
                           const float* __restrict__ g, const float* __restrict__ b,
                           int C, long total) {
    long i = (long)blockIdx.x * blockDim.x + threadIdx.x;
    if (i >= total) return;
    int c = (int)(i % C);
    float v = bf2f(x[i]);
    v = g[c] * (v - stats[c]) * stats[C + c] + b[c];
    x[i] = f2bf(lrelu(v));
}

// ---------------------------------------------------------------------------
// Generic batched WMMA GEMM, 2x2 register-blocked (32x32 macro-tile per wave):
//   C = scale * (A[MxK] * B) + bias
//   A: bf16 row-major (lda), rows clamped for ragged M
//   B layout 0: B(k,n) = Bm[n*ldb + k]   (weight (N,K) row-major / K^T)
//   B layout 1: B(k,n) = Bm[k*ldb + n]   (row-major KxN gather, k < Kreal)
//   out: bf16 or f32 (ldc); ragged rows/cols masked on store
// ---------------------------------------------------------------------------
__global__ void k_gemm_wmma(const unsigned short* __restrict__ A,
                            const unsigned short* __restrict__ Bm,
                            const float* __restrict__ bias,
                            void* __restrict__ Cp, int outF32,
                            int M, int N, int Kloop, int Kreal,
                            int lda, int ldb, int ldc,
                            long sA, long sB, long sC,
                            int bLayout, float scale) {
    int batch = blockIdx.y;
    const unsigned short* Ab = A + (long)batch * sA;
    const unsigned short* Bb = Bm + (long)batch * sB;
    int nt2 = (N + 31) >> 5;
    int mt2 = (M + 31) >> 5;
    int total = mt2 * nt2;
    int wid = blockIdx.x * (blockDim.x >> 5) + (threadIdx.x >> 5);
    if (wid >= total) return;
    int lane = threadIdx.x & 31;
    int lh = lane >> 4, l16 = lane & 15;
    int m0 = (wid / nt2) << 5;
    int n0 = (wid % nt2) << 5;
    int arow0 = m0 + l16;      if (arow0 >= M) arow0 = M - 1;
    int arow1 = m0 + 16 + l16; if (arow1 >= M) arow1 = M - 1;
    int bcol0 = n0 + l16;
    int bcol1 = n0 + 16 + l16;
    int bcol0L = bcol0 < N ? bcol0 : N - 1;
    int bcol1L = bcol1 < N ? bcol1 : N - 1;

    v8f acc00 = {}, acc01 = {}, acc10 = {}, acc11 = {};
    for (int k0 = 0; k0 < Kloop; k0 += 32) {
        v16bf a0 = loadA(Ab, arow0, lda, k0, lh);
        v16bf a1 = loadA(Ab, arow1, lda, k0, lh);
        v16bf b0, b1;
        if (bLayout == 0) {
            b0 = loadB0(Bb, bcol0L, ldb, k0, lh);
            b1 = loadB0(Bb, bcol1L, ldb, k0, lh);
        } else {
            int kb = k0 + (lh << 4);
            #pragma unroll
            for (int i = 0; i < 16; ++i) {
                int k = kb + i;
                b0[i] = (k < Kreal) ? us2bf(Bb[(long)k * ldb + bcol0L]) : us2bf(0);
                b1[i] = (k < Kreal) ? us2bf(Bb[(long)k * ldb + bcol1L]) : us2bf(0);
            }
        }
        acc00 = __builtin_amdgcn_wmma_f32_16x16x32_bf16(false, a0, false, b0, (short)0, acc00, false, false);
        acc01 = __builtin_amdgcn_wmma_f32_16x16x32_bf16(false, a0, false, b1, (short)0, acc01, false, false);
        acc10 = __builtin_amdgcn_wmma_f32_16x16x32_bf16(false, a1, false, b0, (short)0, acc10, false, false);
        acc11 = __builtin_amdgcn_wmma_f32_16x16x32_bf16(false, a1, false, b1, (short)0, acc11, false, false);
    }
    float bv0 = (bias && bcol0 < N) ? bias[bcol0] : 0.f;
    float bv1 = (bias && bcol1 < N) ? bias[bcol1] : 0.f;
    #pragma unroll
    for (int mi = 0; mi < 2; ++mi) {
        #pragma unroll
        for (int r = 0; r < 8; ++r) {
            int row = m0 + mi * 16 + r + (lh << 3);
            if (row >= M) continue;
            long cb = (long)batch * sC + (long)row * ldc;
            float v0 = (mi ? acc10[r] : acc00[r]) * scale + bv0;
            float v1 = (mi ? acc11[r] : acc01[r]) * scale + bv1;
            if (bcol0 < N) {
                if (outF32) ((float*)Cp)[cb + bcol0] = v0;
                else ((unsigned short*)Cp)[cb + bcol0] = f2bf(v0);
            }
            if (bcol1 < N) {
                if (outF32) ((float*)Cp)[cb + bcol1] = v1;
                else ((unsigned short*)Cp)[cb + bcol1] = f2bf(v1);
            }
        }
    }
}

// ---------------------------------------------------------------------------
// Softmax per row of scores (B*T rows, 129 cols, fp32) -> bf16 probs padded to 160
// ---------------------------------------------------------------------------
__global__ void k_softmax(const float* __restrict__ scores, unsigned short* __restrict__ probs) {
    int r = blockIdx.x;                       // 0..4127
    const float* srow = scores + (long)r * 129;
    unsigned short* prow = probs + (long)r * 160;
    __shared__ float red[128];
    int tid = threadIdx.x;
    float v0 = srow[tid];
    float v1 = (tid == 0) ? srow[128] : -3.4e38f;
    red[tid] = fmaxf(v0, v1);
    __syncthreads();
    for (int off = 64; off > 0; off >>= 1) {
        if (tid < off) red[tid] = fmaxf(red[tid], red[tid + off]);
        __syncthreads();
    }
    float mx = red[0];
    __syncthreads();
    float e0 = expf(v0 - mx);
    float e1 = (tid == 0) ? expf(v1 - mx) : 0.f;
    red[tid] = e0 + e1;
    __syncthreads();
    for (int off = 64; off > 0; off >>= 1) {
        if (tid < off) red[tid] += red[tid + off];
        __syncthreads();
    }
    float inv = 1.0f / red[0];
    prow[tid] = f2bf(e0 * inv);
    if (tid == 0) prow[128] = f2bf(e1 * inv);
    if (tid < 31) prow[129 + tid] = 0;        // zero pad K 129..159
}

// z (B,T,512) bf16 -> zt (T,B,512) bf16
__global__ void k_zt(const unsigned short* __restrict__ z, unsigned short* __restrict__ zt) {
    long total = (long)129 * 32 * 512;
    long i = (long)blockIdx.x * blockDim.x + threadIdx.x;
    if (i >= total) return;
    int c = (int)(i & 511);
    long t = i >> 9;
    int b = (int)(t & 31);
    int tt = (int)(t >> 5);
    zt[i] = z[((long)b * 129 + tt) * 512 + c];
}

// ---------------------------------------------------------------------------
// Persistent bidirectional LSTM scan. grid = 32 blocks x 256 threads.
// 256 waves = 2 dirs x (2 M-tiles x 64 N-tiles) of the 32x1024 gate GEMM.
// Atomic grid barrier between GEMM phase and elementwise phase each step.
// ---------------------------------------------------------------------------
static __device__ __forceinline__ void gbar(unsigned* cnt, unsigned* gen, unsigned nb) {
    __threadfence();
    __syncthreads();
    if (threadIdx.x == 0) {
        unsigned g = atomicAdd(gen, 0u);
        if (atomicAdd(cnt, 1u) == nb - 1u) {
            atomicExch(cnt, 0u);
            atomicAdd(gen, 1u);
        } else {
            while (atomicAdd(gen, 0u) == g) { __builtin_amdgcn_s_sleep(2); }
        }
    }
    __syncthreads();
    __threadfence();
}

__global__ void k_lstm_scan(const float* __restrict__ xg_f, const float* __restrict__ xg_b,
                            const unsigned short* __restrict__ whh_f,
                            const unsigned short* __restrict__ whh_b,
                            unsigned short* __restrict__ h_f, unsigned short* __restrict__ h_b,
                            float* __restrict__ c_f, float* __restrict__ c_b,
                            float* __restrict__ gbuf, unsigned short* __restrict__ hs,
                            int T, unsigned* __restrict__ syncCnt, unsigned* __restrict__ syncGen) {
    const unsigned NB = 32;
    int gw = blockIdx.x * 8 + (threadIdx.x >> 5);
    int dir = gw >> 7;
    int tile = gw & 127;
    int m0 = (tile >> 6) << 4;            // batch rows (0 / 16)
    int n0 = (tile & 63) << 4;            // gate cols
    int lane = threadIdx.x & 31;
    int lh = lane >> 4, l16 = lane & 15;
    const float* xg = dir ? xg_b : xg_f;
    const unsigned short* whh = dir ? whh_b : whh_f;
    const unsigned short* h = dir ? h_b : h_f;
    float* gb = gbuf + dir * (32 * 1024);
    int tid = blockIdx.x * blockDim.x + threadIdx.x;
    int col = n0 + l16;

    for (int s = 0; s < T; ++s) {
        int t = dir ? (T - 1 - s) : s;
        v8f acc;
        #pragma unroll
        for (int r = 0; r < 8; ++r) {
            int row = m0 + r + (lh << 3);
            acc[r] = xg[((long)t * 32 + row) * 1024 + col];
        }
        for (int k0 = 0; k0 < 256; k0 += 32) {
            v16bf a = loadA(h, m0 + l16, 256, k0, lh);
            v16bf b = loadB0(whh, col, 256, k0, lh);
            acc = __builtin_amdgcn_wmma_f32_16x16x32_bf16(false, a, false, b,
                                                          (short)0, acc, false, false);
        }
        #pragma unroll
        for (int r = 0; r < 8; ++r) {
            int row = m0 + r + (lh << 3);
            gb[row * 1024 + col] = acc[r];
        }
        gbar(syncCnt, syncGen, NB);

        // elementwise gate phase: 2*32*256 items over 8192 threads
        #pragma unroll
        for (int q = 0; q < 2; ++q) {
            int item = tid + q * 8192;
            int d2 = item >> 13;
            int rem = item & 8191;
            int b2 = rem >> 8;
            int j = rem & 255;
            const float* gB = gbuf + d2 * (32 * 1024) + b2 * 1024;
            float gi = gB[j], gf2 = gB[256 + j], gg = gB[512 + j], go = gB[768 + j];
            float* cc = d2 ? c_b : c_f;
            unsigned short* hh = d2 ? h_b : h_f;
            int tt = d2 ? (T - 1 - s) : s;
            float cn = sigm(gf2) * cc[b2 * 256 + j] + sigm(gi) * tanhf(gg);
            float hn = sigm(go) * tanhf(cn);
            cc[b2 * 256 + j] = cn;
            hh[b2 * 256 + j] = f2bf(hn);
            hs[((long)tt * 32 + b2) * 512 + d2 * 256 + j] = f2bf(hn);
        }
        gbar(syncCnt, syncGen, NB);
    }
}

// ---------------------------------------------------------------------------
// Host side
// ---------------------------------------------------------------------------
typedef unsigned short us;

static inline void launch_gemm(hipStream_t s, const us* A, const us* B, const float* bias,
                               void* C, int outF32, int M, int N, int Kloop, int Kreal,
                               int lda, int ldb, int ldc, int batch,
                               long sA, long sB, long sC, int bLayout, float scale) {
    int mt2 = (M + 31) / 32, nt2 = (N + 31) / 32;
    int tiles = mt2 * nt2;
    dim3 grid((tiles + 7) / 8, batch);
    k_gemm_wmma<<<grid, 256, 0, s>>>(A, B, bias, C, outF32, M, N, Kloop, Kreal,
                                     lda, ldb, ldc, sA, sB, sC, bLayout, scale);
}

extern "C" void kernel_launch(void* const* d_in, const int* in_sizes, int n_in,
                              void* d_out, int out_size, void* d_ws, size_t ws_size,
                              hipStream_t stream) {
    (void)in_sizes; (void)n_in; (void)out_size; (void)ws_size;
    const int T = 129, B = 32;
    uint8_t* ws = (uint8_t*)d_ws;
    size_t cur = 0;
    auto take = [&](size_t bytes) -> uint8_t* {
        uint8_t* p = ws + cur;
        cur = (cur + bytes + 255) & ~(size_t)255;
        return p;
    };

    // --- input pointers ---
    const float* x = (const float*)d_in[0];
    const float* conv_w[7]; const float* conv_b[7];
    for (int l = 0; l < 7; ++l) { conv_w[l] = (const float*)d_in[1 + l]; conv_b[l] = (const float*)d_in[8 + l]; }
    const float* bn_g[3]; const float* bn_b[3];
    for (int i = 0; i < 3; ++i) { bn_g[i] = (const float*)d_in[15 + i]; bn_b[i] = (const float*)d_in[18 + i]; }
    const float* wq = (const float*)d_in[21]; const float* bq = (const float*)d_in[22];
    const float* wk = (const float*)d_in[23]; const float* bk = (const float*)d_in[24];
    const float* wv = (const float*)d_in[25]; const float* bv = (const float*)d_in[26];
    // lstm layer1: 27..36, layer2: 37..46
    const float* wih1f = (const float*)d_in[27]; const float* whh1f = (const float*)d_in[28];
    const float* bih1f = (const float*)d_in[29]; const float* bhh1f = (const float*)d_in[30];
    const float* wih1b = (const float*)d_in[31]; const float* whh1b = (const float*)d_in[32];
    const float* bih1b = (const float*)d_in[33]; const float* bhh1b = (const float*)d_in[34];
    const float* outw1 = (const float*)d_in[35]; const float* outb1 = (const float*)d_in[36];
    const float* wih2f = (const float*)d_in[37]; const float* whh2f = (const float*)d_in[38];
    const float* bih2f = (const float*)d_in[39]; const float* bhh2f = (const float*)d_in[40];
    const float* wih2b = (const float*)d_in[41]; const float* whh2b = (const float*)d_in[42];
    const float* bih2b = (const float*)d_in[43]; const float* bhh2b = (const float*)d_in[44];
    const float* outw2 = (const float*)d_in[45]; const float* outb2 = (const float*)d_in[46];

    // --- workspace arena ---
    const int cCout[7] = {64, 128, 256, 256, 512, 512, 512};
    const int cCin[7]  = {1, 64, 128, 256, 256, 512, 512};
    const int cK[7]    = {3, 3, 3, 3, 3, 3, 2};
    us* wc[7];
    for (int l = 0; l < 7; ++l)
        wc[l] = (us*)take((size_t)cCout[l] * cCin[l] * cK[l] * cK[l] * 2);
    us* wqb = (us*)take(512 * 512 * 2);
    us* wkb = (us*)take(512 * 512 * 2);
    us* wvb = (us*)take(512 * 512 * 2);
    us* wih1fb = (us*)take(1024 * 512 * 2); us* whh1fb = (us*)take(1024 * 256 * 2);
    us* wih1bb = (us*)take(1024 * 512 * 2); us* whh1bb = (us*)take(1024 * 256 * 2);
    us* outw1b = (us*)take(256 * 512 * 2);
    us* wih2fb = (us*)take(1024 * 256 * 2); us* whh2fb = (us*)take(1024 * 256 * 2);
    us* wih2bb = (us*)take(1024 * 256 * 2); us* whh2bb = (us*)take(1024 * 256 * 2);
    us* outw2b = (us*)take(37 * 512 * 2);
    float* cb1f = (float*)take(1024 * 4); float* cb1b = (float*)take(1024 * 4);
    float* cb2f = (float*)take(1024 * 4); float* cb2b = (float*)take(1024 * 4);
    float* bnstats = (float*)take(2 * 512 * 4);
    unsigned* syncblk = (unsigned*)take(256);
    uint8_t* hcblk = take(98304);
    us* h_f = (us*)hcblk;
    us* h_b = (us*)(hcblk + 16384);
    float* c_f = (float*)(hcblk + 32768);
    float* c_b = (float*)(hcblk + 65536);
    float* gbuf = (float*)take(2 * 32 * 1024 * 4);
    uint8_t* ACT0 = take(67108864);
    uint8_t* ACT1 = take(16908288);

    // ACT sub-buffers (ping-pong reuse)
    us* a1 = (us*)ACT0;                       // conv1 out (32,32,512,64)
    us* p1 = (us*)ACT1;
    us* a2 = (us*)ACT0;
    us* p2 = (us*)ACT1;
    us* a3 = (us*)ACT0;
    us* a4 = (us*)ACT1;
    us* p4 = (us*)ACT0;
    us* a5 = (us*)ACT1;
    us* a6 = (us*)ACT0;
    us* p6 = (us*)ACT1;
    us* yb = (us*)ACT0;                       // (B,T,512) bf16
    us* qb = (us*)ACT1;
    us* kb = (us*)(ACT1 + 4227072);
    us* vb = (us*)(ACT1 + 8454144);
    float* sc = (float*)(ACT1 + 12681216);    // scores fp32 (32,129,129)
    us* pr = (us*)(ACT1 + 14811264);          // probs bf16 padded (4128,160)
    us* zb = (us*)(ACT0 + 4227072);           // z (B,T,512)
    us* zt = (us*)(ACT0 + 8454144);           // zt (T,B,512)
    float* xgf = (float*)(ACT0 + 12681216);   // (T,B,1024) fp32
    float* xgb = (float*)(ACT0 + 29589504);
    us* hs = (us*)ACT1;                       // (T,B,512) bf16
    us* pj1 = (us*)(ACT1 + 4227072);          // (T,B,256) bf16

    #define GRID1(n) dim3(((long)(n) + 255) / 256), dim3(256)

    // ---- weight preparation (idempotent each call) ----
    for (int l = 1; l < 7; ++l) {
        long n = (long)cCout[l] * cCin[l] * cK[l] * cK[l];
        k_conv_repack<<<GRID1(n), 0, stream>>>(conv_w[l], wc[l], cCout[l], cCin[l], cK[l], cK[l]);
    }
    k_f32_to_bf16<<<GRID1(512 * 512), 0, stream>>>(wq, wqb, 512 * 512);
    k_f32_to_bf16<<<GRID1(512 * 512), 0, stream>>>(wk, wkb, 512 * 512);
    k_f32_to_bf16<<<GRID1(512 * 512), 0, stream>>>(wv, wvb, 512 * 512);
    k_f32_to_bf16<<<GRID1(1024 * 512), 0, stream>>>(wih1f, wih1fb, 1024 * 512);
    k_f32_to_bf16<<<GRID1(1024 * 256), 0, stream>>>(whh1f, whh1fb, 1024 * 256);
    k_f32_to_bf16<<<GRID1(1024 * 512), 0, stream>>>(wih1b, wih1bb, 1024 * 512);
    k_f32_to_bf16<<<GRID1(1024 * 256), 0, stream>>>(whh1b, whh1bb, 1024 * 256);
    k_f32_to_bf16<<<GRID1(256 * 512), 0, stream>>>(outw1, outw1b, 256 * 512);
    k_f32_to_bf16<<<GRID1(1024 * 256), 0, stream>>>(wih2f, wih2fb, 1024 * 256);
    k_f32_to_bf16<<<GRID1(1024 * 256), 0, stream>>>(whh2f, whh2fb, 1024 * 256);
    k_f32_to_bf16<<<GRID1(1024 * 256), 0, stream>>>(wih2b, wih2bb, 1024 * 256);
    k_f32_to_bf16<<<GRID1(1024 * 256), 0, stream>>>(whh2b, whh2bb, 1024 * 256);
    k_f32_to_bf16<<<GRID1(37 * 512), 0, stream>>>(outw2, outw2b, 37 * 512);
    k_add_bias<<<4, 256, 0, stream>>>(bih1f, bhh1f, cb1f, 1024);
    k_add_bias<<<4, 256, 0, stream>>>(bih1b, bhh1b, cb1b, 1024);
    k_add_bias<<<4, 256, 0, stream>>>(bih2f, bhh2f, cb2f, 1024);
    k_add_bias<<<4, 256, 0, stream>>>(bih2b, bhh2b, cb2b, 1024);
    (void)hipMemsetAsync(syncblk, 0, 256, stream);

    // ---- conv trunk ----
    k_conv1<<<GRID1((long)32 * 32 * 512 * 64), 0, stream>>>(x, conv_w[0], conv_b[0], a1);
    k_maxpool<<<GRID1((long)32 * 16 * 256 * 64), 0, stream>>>(a1, p1, 32, 32, 512, 64, 16, 256, 2, 2, 0);

    auto conv = [&](const us* in, int l, int Hi, int Wi, int Ho, int Wo, int pad, int act, us* out) {
        int tiles = ((32 * Ho * Wo) / 32) * (cCout[l] / 32);
        k_conv_wmma<<<dim3((tiles + 7) / 8), dim3(256), 0, stream>>>(
            in, wc[l], conv_b[l], out, 32, Hi, Wi, cCin[l], Ho, Wo, cCout[l], cK[l], cK[l], pad, act);
    };

    conv(p1, 1, 16, 256, 16, 256, 1, 1, a2);
    k_maxpool<<<GRID1((long)32 * 8 * 128 * 128), 0, stream>>>(a2, p2, 32, 16, 256, 128, 8, 128, 2, 2, 0);
    conv(p2, 2, 8, 128, 8, 128, 1, 0, a3);
    k_bn_stats<<<256, 256, 0, stream>>>(a3, 256, 32 * 8 * 128, bnstats);
    k_bn_apply<<<GRID1((long)32 * 8 * 128 * 256), 0, stream>>>(a3, bnstats, bn_g[0], bn_b[0], 256, (long)32 * 8 * 128 * 256);
    conv(a3, 3, 8, 128, 8, 128, 1, 1, a4);
    k_maxpool<<<GRID1((long)32 * 4 * 129 * 256), 0, stream>>>(a4, p4, 32, 8, 128, 256, 4, 129, 2, 1, 1);
    conv(p4, 4, 4, 129, 4, 129, 1, 0, a5);
    k_bn_stats<<<512, 256, 0, stream>>>(a5, 512, 32 * 4 * 129, bnstats);
    k_bn_apply<<<GRID1((long)32 * 4 * 129 * 512), 0, stream>>>(a5, bnstats, bn_g[1], bn_b[1], 512, (long)32 * 4 * 129 * 512);
    conv(a5, 5, 4, 129, 4, 129, 1, 1, a6);
    k_maxpool<<<GRID1((long)32 * 2 * 130 * 512), 0, stream>>>(a6, p6, 32, 4, 129, 512, 2, 130, 2, 1, 1);
    conv(p6, 6, 2, 130, 1, 129, 0, 0, yb);
    k_bn_stats<<<512, 256, 0, stream>>>(yb, 512, 32 * 1 * 129, bnstats);
    k_bn_apply<<<GRID1((long)32 * 129 * 512), 0, stream>>>(yb, bnstats, bn_g[2], bn_b[2], 512, (long)32 * 129 * 512);

    // ---- attention ----
    const int MT = B * T;   // 4128
    launch_gemm(stream, yb, wqb, bq, qb, 0, MT, 512, 512, 512, 512, 512, 512, 1, 0, 0, 0, 0, 1.f);
    launch_gemm(stream, yb, wkb, bk, kb, 0, MT, 512, 512, 512, 512, 512, 512, 1, 0, 0, 0, 0, 1.f);
    launch_gemm(stream, yb, wvb, bv, vb, 0, MT, 512, 512, 512, 512, 512, 512, 1, 0, 0, 0, 0, 1.f);
    launch_gemm(stream, qb, kb, nullptr, sc, 1, T, T, 512, 512, 512, 512, T, B,
                (long)T * 512, (long)T * 512, (long)T * T, 0, 0.044194173824159216f);
    k_softmax<<<MT, 128, 0, stream>>>(sc, pr);
    launch_gemm(stream, pr, vb, nullptr, zb, 0, T, 512, 160, T, 160, 512, 512, B,
                (long)T * 160, (long)T * 512, (long)T * 512, 1, 1.f);
    k_zt<<<GRID1((long)T * B * 512), 0, stream>>>(zb, zt);

    // ---- BiLSTM layer 1 ----
    launch_gemm(stream, zt, wih1fb, cb1f, xgf, 1, MT, 1024, 512, 512, 512, 512, 1024, 1, 0, 0, 0, 0, 1.f);
    launch_gemm(stream, zt, wih1bb, cb1b, xgb, 1, MT, 1024, 512, 512, 512, 512, 1024, 1, 0, 0, 0, 0, 1.f);
    (void)hipMemsetAsync(hcblk, 0, 98304, stream);
    k_lstm_scan<<<32, 256, 0, stream>>>(xgf, xgb, whh1fb, whh1bb, h_f, h_b, c_f, c_b,
                                        gbuf, hs, T, syncblk, syncblk + 64);
    launch_gemm(stream, hs, outw1b, outb1, pj1, 0, MT, 256, 512, 512, 512, 512, 256, 1, 0, 0, 0, 0, 1.f);

    // ---- BiLSTM layer 2 ----
    launch_gemm(stream, pj1, wih2fb, cb2f, xgf, 1, MT, 1024, 256, 256, 256, 256, 1024, 1, 0, 0, 0, 0, 1.f);
    launch_gemm(stream, pj1, wih2bb, cb2b, xgb, 1, MT, 1024, 256, 256, 256, 256, 1024, 1, 0, 0, 0, 0, 1.f);
    (void)hipMemsetAsync(hcblk, 0, 98304, stream);
    k_lstm_scan<<<32, 256, 0, stream>>>(xgf, xgb, whh2fb, whh2bb, h_f, h_b, c_f, c_b,
                                        gbuf, hs, T, syncblk, syncblk + 64);
    launch_gemm(stream, hs, outw2b, outb2, d_out, 1, MT, 37, 512, 512, 512, 512, 37, 1, 0, 0, 0, 0, 1.f);
}
